// MultiHeadAttention_60172491817269
// MI455X (gfx1250) — matrix operand order repaired and checked by reading, not compiled, not run on hardware
//
#include <hip/hip_runtime.h>

// ============================================================================
// MI455X (gfx1250) attention forward: Q/K/V proj + RoPE + QK^T + causal mask
// + softmax + attn@V + out proj.  All GEMMs run on v_wmma_f32_16x16x32_bf16
// (f32 accumulate).  Global->LDS staging uses gfx1250 async-to-LDS DMA
// (ASYNCcnt) with LDS double buffering so HBM latency hides under WMMA.
// attn (post-softmax) is materialized in d_out as required.
// Workspace use: ~80 MB.
// ============================================================================

typedef __bf16 bf16_t;
typedef __attribute__((ext_vector_type(16))) bf16_t v16bf;
typedef __attribute__((ext_vector_type(8)))  bf16_t v8bf;
typedef __attribute__((ext_vector_type(4)))  bf16_t v4bf;
typedef __attribute__((ext_vector_type(8)))  float  v8f;

#define BM 128
#define BN 128
#define BK 32
#define LDSTR 40  // BK + 8 bf16 pad -> 80B row pitch: 16B aligned, bank-spread

// ---- gfx1250 async global->LDS DMA (16B per lane), ASYNCcnt tracked --------
// Generic flat address of a __shared__ object carries the LDS byte offset in
// bits [31:0] (ISA 10.2).
__device__ __forceinline__ void async_ld16(void* lds, const void* gaddr) {
  asm volatile("global_load_async_to_lds_b128 %0, %1, off"
               :: "v"((unsigned)(unsigned long long)lds), "v"(gaddr)
               : "memory");
}
__device__ __forceinline__ void wait_async0() {
  asm volatile("s_wait_asynccnt 0x0" ::: "memory");
}

// ---------------------------------------------------------------------------
// Common 128x128 WMMA tile:  C = scale * (A[M,K] @ B[N,K]^T)   (K contiguous
// in both operands).  8 waves, each owns a 32x64 macro-tile (2x4 wmma tiles).
// TA = float (converted to bf16, register-pipelined) or bf16 (async DMA).
// ---------------------------------------------------------------------------
template<typename TA, bool BF16_OUT, bool TRANS_OUT, bool CAUSAL>
__device__ __forceinline__ void gemm_block_device(
    const TA* __restrict__ A, long lda,
    const bf16_t* __restrict__ B, long ldb,
    void* __restrict__ Cout, long ldc,
    int K, float scale, int m0, int n0,
    int transS, long transBatch)
{
  __shared__ bf16_t As[2][BM * LDSTR];
  __shared__ bf16_t Bs[2][BN * LDSTR];

  const int tid  = threadIdx.x;
  const int wave = tid >> 5;
  const int lane = tid & 31;
  const int hl   = lane >> 4;   // lane half (ISA wave32 fragment layout)
  const int l16  = lane & 15;
  const int wm   = (wave >> 1) * 32;  // wave m-offset in block
  const int wn   = (wave & 1) * 64;   // wave n-offset in block

  // staging coordinates: bf16 tiles = 512 x 16B segs, 2 per thread
  const int srow = tid >> 2;          // 0..63 (+64 on second seg)
  const int sks  = (tid & 3) * 8;     // bf16 elems, 16B aligned
  // f32 A tiles = 1024 x 16B segs, 4 per thread
  const int frow = tid >> 3;          // 0..31 (+32*i)
  const int fks  = (tid & 7) * 4;

  constexpr bool A_BF16 = (sizeof(TA) == 2);

  auto stageA_async = [&](int p, int k0) {
#pragma unroll
    for (int i = 0; i < 2; ++i) {
      const int row = srow + i * 64;
      async_ld16(&As[p][row * LDSTR + sks],
                 (const bf16_t*)A + (long)(m0 + row) * lda + (k0 + sks));
    }
  };
  auto stageB_async = [&](int p, int k0) {
#pragma unroll
    for (int i = 0; i < 2; ++i) {
      const int row = srow + i * 64;
      async_ld16(&Bs[p][row * LDSTR + sks],
                 B + (long)(n0 + row) * ldb + (k0 + sks));
    }
  };

  v8f acc[2][4] = {};

  // ---- preamble: fill buffer 0 ----
  if constexpr (A_BF16) {
    stageA_async(0, 0);
  } else {
    float4 f[4];
#pragma unroll
    for (int i = 0; i < 4; ++i)
      f[i] = *(const float4*)(A + (long)(m0 + frow + i * 32) * lda + fks);
#pragma unroll
    for (int i = 0; i < 4; ++i) {
      bf16_t* d = &As[0][(frow + i * 32) * LDSTR + fks];
      d[0] = (bf16_t)f[i].x; d[1] = (bf16_t)f[i].y;
      d[2] = (bf16_t)f[i].z; d[3] = (bf16_t)f[i].w;
    }
  }
  stageB_async(0, 0);
  wait_async0();
  __syncthreads();

  int p = 0;
  for (int k0 = 0; k0 < K; k0 += BK, p ^= 1) {
    const bool next = (k0 + BK) < K;

    // ---- kick off next tile (DMA / register loads) before the math ----
    float4 fnext[4];
    if (next) {
      if constexpr (A_BF16) {
        stageA_async(p ^ 1, k0 + BK);
      } else {
#pragma unroll
        for (int i = 0; i < 4; ++i)
          fnext[i] = *(const float4*)(A + (long)(m0 + frow + i * 32) * lda +
                                      (k0 + BK + fks));
      }
      stageB_async(p ^ 1, k0 + BK);
    }
    if (k0 + 2 * BK < K) {  // keep L2 warm two tiles ahead (global_prefetch_b8)
      __builtin_prefetch((const char*)A +
                         ((long)(m0 + (tid >> 1)) * lda + (k0 + 2 * BK)) * sizeof(TA), 0, 3);
      __builtin_prefetch(B + (long)(n0 + (tid >> 1)) * ldb + (k0 + 2 * BK), 0, 3);
    }

    // ---- assemble wave32 WMMA fragments (ISA 7.12.2 layouts) ----
    v16bf af[2], bfv[4];
    const int ka = hl * 8;   // A: lanes0-15 K{0..7,16..23}; lanes16-31 K{8..15,24..31}
#pragma unroll
    for (int i = 0; i < 2; ++i) {
      const bf16_t* r = &As[p][(wm + i * 16 + l16) * LDSTR];
      v8bf lo = *(const v8bf*)&r[ka];
      v8bf hi = *(const v8bf*)&r[ka + 16];
      af[i] = __builtin_shufflevector(lo, hi, 0,1,2,3,4,5,6,7,8,9,10,11,12,13,14,15);
    }
    const int kb = hl * 8;   // B: lanes0-15 K{0..7,16..23}; lanes16-31 K{8..15,24..31}
#pragma unroll
    for (int j = 0; j < 4; ++j) {
      const bf16_t* r = &Bs[p][(wn + j * 16 + l16) * LDSTR];
      v8bf lo = *(const v8bf*)&r[kb];
      v8bf hi = *(const v8bf*)&r[kb + 16];
      bfv[j] = __builtin_shufflevector(lo, hi, 0,1,2,3,4,5,6,7,8,9,10,11,12,13,14,15);
    }
#pragma unroll
    for (int i = 0; i < 2; ++i)
#pragma unroll
      for (int j = 0; j < 4; ++j)
        acc[i][j] = __builtin_amdgcn_wmma_f32_16x16x32_bf16(
            false, af[i], false, bfv[j], (short)0, acc[i][j], false, false);

    if (next) {
      if constexpr (!A_BF16) {
        // f32 path: convert + LDS-store the prefetched registers after the math
#pragma unroll
        for (int i = 0; i < 4; ++i) {
          bf16_t* d = &As[p ^ 1][(frow + i * 32) * LDSTR + fks];
          d[0] = (bf16_t)fnext[i].x; d[1] = (bf16_t)fnext[i].y;
          d[2] = (bf16_t)fnext[i].z; d[3] = (bf16_t)fnext[i].w;
        }
      }
      wait_async0();     // our async DMA into buffer p^1 done
      __syncthreads();   // everyone's DMA/ds-stores done, everyone done reading p
    }
  }

  // ---- epilogue: C/D layout = VGPR r -> row (hl*8 + r), col = l16 ----
#pragma unroll
  for (int i = 0; i < 2; ++i) {
    const int gmb = m0 + wm + i * 16 + hl * 8;
#pragma unroll
    for (int j = 0; j < 4; ++j) {
      const int gn = n0 + wn + j * 16 + l16;
      if constexpr (TRANS_OUT) {
        // transposed bf16 store: 8 consecutive rows -> one 16B write
        v8bf t;
#pragma unroll
        for (int r = 0; r < 8; ++r) t[r] = (bf16_t)(acc[i][j][r] * scale);
        const int b = gmb / transS;
        const int s = gmb - b * transS;
        bf16_t* C = (bf16_t*)Cout;
        *(v8bf*)&C[(long)b * transBatch + (long)gn * transS + s] = t;
      } else if constexpr (BF16_OUT) {
        bf16_t* C = (bf16_t*)Cout;
#pragma unroll
        for (int r = 0; r < 8; ++r)
          C[(long)(gmb + r) * ldc + gn] = (bf16_t)(acc[i][j][r] * scale);
      } else {
        float* C = (float*)Cout;
#pragma unroll
        for (int r = 0; r < 8; ++r) {
          const int gm = gmb + r;
          float v = acc[i][j][r] * scale;
          if constexpr (CAUSAL) { if (gn > gm) v = -1.0e9f; }
          C[(long)gm * ldc + gn] = v;
        }
      }
    }
  }
}

// ---------------------------------------------------------------------------
// Kernel wrappers
// ---------------------------------------------------------------------------
template<typename TA, bool BF16_OUT>
__global__ void __launch_bounds__(256) gemm_plain(
    const TA* A, long lda, const bf16_t* B, long ldb,
    void* C, long ldc, int K, float scale)
{
  gemm_block_device<TA, BF16_OUT, false, false>(
      A, lda, B, ldb, C, ldc, K, scale, blockIdx.y * BM, blockIdx.x * BN, 0, 0);
}

// V projection with transposed store: v^T laid out [b][kv*D + d][s]
__global__ void __launch_bounds__(256) gemm_vproj(
    const bf16_t* A, long lda, const bf16_t* B, long ldb,
    bf16_t* C, int K, int transS, long transBatch)
{
  gemm_block_device<bf16_t, true, true, false>(
      A, lda, B, ldb, C, 0, K, 1.0f, blockIdx.y * BM, blockIdx.x * BN,
      transS, transBatch);
}

// scores = scale * Q K^T with causal mask, per (b,h)
__global__ void __launch_bounds__(256) gemm_scores_k(
    const bf16_t* q, const bf16_t* k, float* attn,
    int S, int H, int KVH, int D, float scale)
{
  const int z = blockIdx.z, b = z / H, h = z % H, kv = h / (H / KVH);
  const bf16_t* A = q + (long)b * S * H * D + (long)h * D;
  const bf16_t* B = k + (long)b * S * KVH * D + (long)kv * D;
  float* C = attn + (long)z * S * S;
  gemm_block_device<bf16_t, false, false, true>(
      A, (long)H * D, B, (long)KVH * D, C, S, D, scale,
      blockIdx.y * BM, blockIdx.x * BN, 0, 0);
}

// out_heads = attn @ V, per (b,h); attn f32 -> bf16 during staging
__global__ void __launch_bounds__(256) gemm_av_k(
    const float* attn, const bf16_t* vT, bf16_t* oh,
    int S, int H, int KVH, int D)
{
  const int z = blockIdx.z, b = z / H, h = z % H, kv = h / (H / KVH);
  const float* A = attn + (long)z * S * S;
  const bf16_t* B = vT + ((long)b * KVH * D + (long)kv * D) * S;
  bf16_t* C = oh + (long)b * S * H * D + (long)h * D;
  gemm_block_device<float, true, false, false>(
      A, S, B, S, C, (long)H * D, S, 1.0f,
      blockIdx.y * BM, blockIdx.x * BN, 0, 0);
}

// ---------------------------------------------------------------------------
__global__ void __launch_bounds__(256) f32_to_bf16_kernel(
    const float* __restrict__ in, bf16_t* __restrict__ out, long n4)
{
  long i = (long)blockIdx.x * 256 + threadIdx.x;
  if (i < n4) {
    float4 f = ((const float4*)in)[i];
    v4bf o;
    o[0] = (bf16_t)f.x; o[1] = (bf16_t)f.y;
    o[2] = (bf16_t)f.z; o[3] = (bf16_t)f.w;
    ((v4bf*)out)[i] = o;
  }
}

// RoPE in place on bf16 [b,s,nh,hd]; one thread per (even,odd) pair
__global__ void __launch_bounds__(256) rope_kernel(
    bf16_t* x, int S, int nh, int hd, long total)
{
  long gid = (long)blockIdx.x * 256 + threadIdx.x;
  if (gid >= total) return;
  const int hp = hd >> 1;
  int  d2 = (int)(gid % hp);
  long t  = gid / hp;
  int  h  = (int)(t % nh); t /= nh;
  int  s  = (int)(t % S);
  long b  = t / S;
  // inv_freq = theta^(-2*d2/hd);  ln(10000) = 9.210340371976184
  float freq = __expf(-(float)(2 * d2) * (9.210340371976184f / (float)hd));
  float sn, cs;
  sincosf((float)s * freq, &sn, &cs);
  long off = (((b * S + s) * nh + h) * (long)hd) + 2 * d2;
  float e = (float)x[off], o = (float)x[off + 1];
  x[off]     = (bf16_t)(e * cs - o * sn);
  x[off + 1] = (bf16_t)(o * cs + e * sn);
}

// Row softmax in place; one block per row of ncols (=2048) floats
__global__ void __launch_bounds__(256) softmax_kernel(float* attn, int ncols)
{
  float* p = attn + (long)blockIdx.x * ncols;
  __shared__ float red[256];
  const int tid = threadIdx.x;
  float vals[8];
  float mx = -3.0e38f;
#pragma unroll
  for (int i = 0; i < 8; ++i) { vals[i] = p[tid + i * 256]; mx = fmaxf(mx, vals[i]); }
  red[tid] = mx; __syncthreads();
  for (int s = 128; s > 0; s >>= 1) {
    if (tid < s) red[tid] = fmaxf(red[tid], red[tid + s]);
    __syncthreads();
  }
  mx = red[0]; __syncthreads();
  float sum = 0.0f;
#pragma unroll
  for (int i = 0; i < 8; ++i) { vals[i] = __expf(vals[i] - mx); sum += vals[i]; }
  red[tid] = sum; __syncthreads();
  for (int s = 128; s > 0; s >>= 1) {
    if (tid < s) red[tid] += red[tid + s];
    __syncthreads();
  }
  const float inv = 1.0f / red[0];
#pragma unroll
  for (int i = 0; i < 8; ++i) p[tid + i * 256] = vals[i] * inv;
}

// ---------------------------------------------------------------------------
extern "C" void kernel_launch(void* const* d_in, const int* in_sizes, int n_in,
                              void* d_out, int out_size, void* d_ws, size_t ws_size,
                              hipStream_t stream) {
  (void)in_sizes; (void)n_in; (void)out_size; (void)ws_size;
  constexpr int  Bb = 2, S = 2048, DM = 2048, H = 16, KVH = 4, D = 128;
  constexpr long MS = (long)Bb * S;  // 4096 rows

  const float* x  = (const float*)d_in[0];
  const float* Wq = (const float*)d_in[1];
  const float* Wk = (const float*)d_in[2];
  const float* Wv = (const float*)d_in[3];
  const float* Wo = (const float*)d_in[4];
  // d_in[5] = attention_mask: causal, handled analytically

  float* out  = (float*)d_out;
  float* attn = out + MS * DM;  // second tuple element, [B,H,S,S] f32

  char* ws = (char*)d_ws;
  bf16_t* xb  = (bf16_t*)ws; ws += MS * DM * 2;
  bf16_t* wqb = (bf16_t*)ws; ws += (long)DM * DM * 2;
  bf16_t* wkb = (bf16_t*)ws; ws += (long)KVH * D * DM * 2;
  bf16_t* wvb = (bf16_t*)ws; ws += (long)KVH * D * DM * 2;
  bf16_t* wob = (bf16_t*)ws; ws += (long)DM * DM * 2;
  bf16_t* qb  = (bf16_t*)ws; ws += MS * DM * 2;        // [b,s,h,d] bf16
  bf16_t* kb  = (bf16_t*)ws; ws += MS * KVH * D * 2;   // [b,s,kv,d] bf16
  bf16_t* vtb = (bf16_t*)ws; ws += MS * KVH * D * 2;   // [b,kv*d,s] bf16 (transposed)
  bf16_t* ohb = (bf16_t*)ws;                           // [b,s,h,d] bf16

  auto cvt = [&](const float* src, bf16_t* dst, long n) {
    long n4 = n / 4;
    f32_to_bf16_kernel<<<dim3((unsigned)((n4 + 255) / 256)), 256, 0, stream>>>(src, dst, n4);
  };
  cvt(x,  xb,  MS * DM);
  cvt(Wq, wqb, (long)DM * DM);
  cvt(Wk, wkb, (long)KVH * D * DM);
  cvt(Wv, wvb, (long)KVH * D * DM);
  cvt(Wo, wob, (long)DM * DM);

  // Q/K/V projections (C = X Wx^T)
  gemm_plain<bf16_t, true><<<dim3(DM / BN, MS / BM), 256, 0, stream>>>(
      xb, DM, wqb, DM, qb, DM, DM, 1.0f);
  gemm_plain<bf16_t, true><<<dim3(KVH * D / BN, MS / BM), 256, 0, stream>>>(
      xb, DM, wkb, DM, kb, KVH * D, DM, 1.0f);
  gemm_vproj<<<dim3(KVH * D / BN, MS / BM), 256, 0, stream>>>(
      xb, DM, wvb, DM, vtb, DM, S, (long)KVH * D * S);

  // RoPE on Q and K
  rope_kernel<<<dim3((unsigned)(MS * H * (D / 2) / 256)), 256, 0, stream>>>(
      qb, S, H, D, MS * H * (D / 2));
  rope_kernel<<<dim3((unsigned)(MS * KVH * (D / 2) / 256)), 256, 0, stream>>>(
      kb, S, KVH, D, MS * KVH * (D / 2));

  // scores = QK^T/sqrt(D) + causal mask, written straight into attn output
  gemm_scores_k<<<dim3(S / BN, S / BM, Bb * H), 256, 0, stream>>>(
      qb, kb, attn, S, H, KVH, D, 0.08838834764831845f);

  // softmax rows in place
  softmax_kernel<<<dim3((unsigned)((long)Bb * H * S)), 256, 0, stream>>>(attn, S);

  // out_heads = attn @ V
  gemm_av_k<<<dim3(1, S / BM, Bb * H), 256, 0, stream>>>(attn, vtb, ohb, S, H, KVH, D);

  // out = out_heads @ Wo^T  (f32 output)
  gemm_plain<bf16_t, false><<<dim3(DM / BN, MS / BM), 256, 0, stream>>>(
      ohb, DM, wob, DM, out, DM, DM, 1.0f);
}
